// LongFormer_55198919688433
// MI455X (gfx1250) — matrix operand order repaired and checked by reading, not compiled, not run on hardware
//
#include <hip/hip_runtime.h>

// Longformer banded attention for MI455X (gfx1250), wave32 + WMMA f16->f32 + TDM.
//   B=2 S=4096 E=768 H=12 D=64  w=256 (one-sided window) C=16 chunks
// Stage 0: X -> f16, W -> f16 transposed (one-time, memory-bound).
// Stage 1: QKV projection GEMM: 4-wave WG computes 64x64 tile; A-tile staged in
//          LDS via tensor_load_to_lds (double buffered, s_wait_tensorcnt).
// Stage 2: flash-style banded attention, one wave per 16-query tile.

#define Bv 2
#define Sv 4096
#define Ev 768
#define Hv 12
#define Dv 64
#define Wc 256      // chunk width (one-sided window w)
#define Cc 16       // number of chunks

typedef __attribute__((ext_vector_type(16))) _Float16 v16h;
typedef __attribute__((ext_vector_type(8)))  _Float16 v8h;
typedef __attribute__((ext_vector_type(8)))  float    v8f;
typedef __attribute__((ext_vector_type(4)))  unsigned u32x4;
typedef __attribute__((ext_vector_type(8)))  int      i32x8;
typedef __attribute__((ext_vector_type(4)))  int      i32x4;
typedef __attribute__((address_space(3)))    _Float16 lds_f16;

union V16 { v16h v; v8h h[2]; };

#if __has_builtin(__builtin_amdgcn_tensor_load_to_lds)
#define HAVE_TDM 1
#else
#define HAVE_TDM 0
#endif

__device__ __forceinline__ v8f wmma_f16(v16h a, v16h b, v8f c) {
  return __builtin_amdgcn_wmma_f32_16x16x32_f16(false, a, false, b, (short)0, c,
                                                false, false);
}

__device__ __forceinline__ void wait_tensorcnt0() {
#if __has_builtin(__builtin_amdgcn_s_wait_tensorcnt)
  __builtin_amdgcn_s_wait_tensorcnt(0);
#else
  asm volatile("s_wait_tensorcnt 0x0" ::: "memory");
#endif
}

// 2D tile load (2-byte elements) global -> LDS via Tensor Data Mover.
// tile_x elements contiguous per row, tile_y rows, row stride = stride_x elems.
__device__ __forceinline__ void tdm_load_2d(unsigned lds_byte_off,
                                            const _Float16* gptr,
                                            unsigned tile_x, unsigned tile_y,
                                            unsigned tensor_x, unsigned tensor_y,
                                            unsigned stride_x) {
#if HAVE_TDM
  const unsigned long long ga = (unsigned long long)gptr;
  u32x4 g0;
  g0.x = 1u;                                             // count=1 (valid user D#)
  g0.y = lds_byte_off;                                   // lds_addr [63:32]
  g0.z = (unsigned)(ga & 0xffffffffu);                   // global_addr [95:64]
  g0.w = (unsigned)((ga >> 32) & 0x01ffffffu) | (2u << 30); // addr[56:32] | type=2
  i32x8 g1;
  g1[0] = (int)(1u << 16);                               // wg_mask=0, data_size=1 (2B)
  g1[1] = (int)((tensor_x & 0xffffu) << 16);             // tensor_dim0[15:0]
  g1[2] = (int)(((tensor_x >> 16) & 0xffffu) | ((tensor_y & 0xffffu) << 16));
  g1[3] = (int)(((tensor_y >> 16) & 0xffffu) | ((tile_x & 0xffffu) << 16));
  g1[4] = (int)(tile_y & 0xffffu);                       // tile_dim1 | tile_dim2=0
  g1[5] = (int)stride_x;                                 // tensor_dim0_stride[31:0]
  g1[6] = 0;                                             // stride0[47:32] | stride1[15:0]
  g1[7] = 0;                                             // stride1[47:16]
  i32x4 z4 = {0, 0, 0, 0};
#if __clang_major__ >= 23
  i32x8 z8 = {0, 0, 0, 0, 0, 0, 0, 0};
  __builtin_amdgcn_tensor_load_to_lds(g0, g1, z4, z4, z8, 0);
#else
  __builtin_amdgcn_tensor_load_to_lds(g0, g1, z4, z4, 0);
#endif
#else
  // Fallback: cooperative wave copy (tile_y rows, one row per lane; tile_y<=32)
  const int lane = threadIdx.x & 31;
  if ((unsigned)lane < tile_y) {
    const _Float16* src = gptr + (size_t)lane * stride_x;
    lds_f16* dst = (lds_f16*)(size_t)(lds_byte_off + lane * tile_x * 2);
    for (unsigned e = 0; e < tile_x; ++e) dst[e] = src[e];
  }
  (void)tensor_x; (void)tensor_y;
#endif
}

// ---------------------------------------------------------------------------
// Stage 0a: X f32 -> f16
// ---------------------------------------------------------------------------
__global__ __launch_bounds__(256)
void cvt_x_kernel(const float* __restrict__ X, _Float16* __restrict__ Xh, int n) {
  const int i = blockIdx.x * 256 + threadIdx.x;
  if (i < n) Xh[i] = (_Float16)X[i];
}

// ---------------------------------------------------------------------------
// Stage 0b: W[k][n] f32 -> WhT[which][n][k] f16 (transposed for B-fragments)
// ---------------------------------------------------------------------------
__global__ __launch_bounds__(256)
void cvt_w_kernel(const float* __restrict__ Wq, const float* __restrict__ Wk,
                  const float* __restrict__ Wv, _Float16* __restrict__ WhT) {
  const int i = blockIdx.x * 256 + threadIdx.x;   // over 3*E*E
  const int which = i / (Ev * Ev);
  const int r = i - which * (Ev * Ev);
  const int k = r / Ev, n = r - k * Ev;           // consecutive threads: n fast
  const float* W = (which == 0) ? Wq : (which == 1) ? Wk : Wv;
  WhT[(size_t)which * Ev * Ev + (size_t)n * Ev + k] = (_Float16)W[(size_t)k * Ev + n];
}

// ---------------------------------------------------------------------------
// Stage 1: GEMM. Workgroup (128 thr, 4 waves) computes a 64x64 tile of one
// projection. A-tile (64x32 f16) double-buffered in LDS via TDM; B from L2.
// grid = (B*S/64, E/64, 3)
// ---------------------------------------------------------------------------
__global__ __launch_bounds__(128)
void qkv_gemm_kernel(const _Float16* __restrict__ Xh,
                     const _Float16* __restrict__ WhT,
                     const float* __restrict__ bq, const float* __restrict__ bk,
                     const float* __restrict__ bv,
                     _Float16* __restrict__ qws, _Float16* __restrict__ kws,
                     _Float16* __restrict__ vtws)
{
  __shared__ _Float16 atile[2][64 * 32];          // [buf][row*32 + k]

  const int lane = threadIdx.x & 31;
  const int wid  = threadIdx.x >> 5;              // 0..3
  const int lo = lane & 15, hi = lane >> 4;
  const int m0 = blockIdx.x << 6;                 // 64 rows over B*S
  const int n0 = blockIdx.y << 6;                 // 64 cols over E
  const int which = blockIdx.z;

  const _Float16* Wt = WhT + (size_t)which * Ev * Ev;
  const float* bb = (which == 0) ? bq : (which == 1) ? bk : bv;

  const int rw = (wid >> 1) * 32;                 // wave's row offset in tile
  const int cw = (wid & 1) * 32;                  // wave's col offset in tile

  unsigned lds_base = (unsigned)(unsigned long long)(lds_f16*)&atile[0][0];

  // prime buffer 0: waves 0/1 each DMA a 32x32 slice
  if (wid < 2)
    tdm_load_2d(lds_base + (unsigned)(wid * 32 * 32 * 2),
                Xh + (size_t)(m0 + wid * 32) * Ev, 32, 32, Ev, Bv * Sv, Ev);

  v8f acc00 = {}, acc01 = {}, acc10 = {}, acc11 = {};
  int cur = 0;
  for (int kk = 0; kk < Ev; kk += 32) {
    wait_tensorcnt0();
    __syncthreads();                              // tile[cur] ready for all waves
    if (kk + 32 < Ev && wid < 2)
      tdm_load_2d(lds_base + (unsigned)((cur ^ 1) * 64 * 32 * 2 + wid * 32 * 32 * 2),
                  Xh + (size_t)(m0 + wid * 32) * Ev + (kk + 32),
                  32, 32, Ev, Bv * Sv, Ev);

    // A fragments (rows rw+mi*16), from LDS
    V16 a0, a1;
    {
      const _Float16* ap0 = &atile[cur][(rw + lo) * 32 + hi * 8];
      a0.h[0] = *(const v8h*)ap0;
      a0.h[1] = *(const v8h*)(ap0 + 16);
      const _Float16* ap1 = &atile[cur][(rw + 16 + lo) * 32 + hi * 8];
      a1.h[0] = *(const v8h*)ap1;
      a1.h[1] = *(const v8h*)(ap1 + 16);
    }
    // B fragments (cols cw+ni*16), direct from WhT
    V16 b0, b1;
    {
      const _Float16* bp0 = Wt + (size_t)(n0 + cw + lo) * Ev + kk + hi * 8;
      b0.h[0] = *(const v8h*)bp0;
      b0.h[1] = *(const v8h*)(bp0 + 16);
      const _Float16* bp1 = Wt + (size_t)(n0 + cw + 16 + lo) * Ev + kk + hi * 8;
      b1.h[0] = *(const v8h*)bp1;
      b1.h[1] = *(const v8h*)(bp1 + 16);
    }
    acc00 = wmma_f16(a0.v, b0.v, acc00);
    acc01 = wmma_f16(a0.v, b1.v, acc01);
    acc10 = wmma_f16(a1.v, b0.v, acc10);
    acc11 = wmma_f16(a1.v, b1.v, acc11);

    __syncthreads();                              // done reading tile[cur]
    cur ^= 1;
  }

  const float scl = (which == 0) ? 0.125f : 1.0f; // q * 1/sqrt(64)
#pragma unroll
  for (int mi = 0; mi < 2; ++mi) {
#pragma unroll
    for (int ni = 0; ni < 2; ++ni) {
      const v8f acc = (mi == 0) ? (ni == 0 ? acc00 : acc01)
                                : (ni == 0 ? acc10 : acc11);
      const int ncol = n0 + cw + ni * 16 + lo;
      const float bias = bb[ncol];
      const int h = ncol >> 6, d = ncol & 63;
#pragma unroll
      for (int r = 0; r < 8; ++r) {
        const int mrow = m0 + rw + mi * 16 + r + hi * 8;
        const int b = mrow >> 12;
        const int s = mrow & (Sv - 1);
        const _Float16 val = (_Float16)((acc[r] + bias) * scl);
        const size_t bh = (size_t)(b * Hv + h);
        if (which == 0)      qws[(bh * Sv + s) * Dv + d] = val;
        else if (which == 1) kws[(bh * Sv + s) * Dv + d] = val;
        else                 vtws[(bh * Dv + d) * Sv + s] = val;
      }
    }
  }
}

// ---------------------------------------------------------------------------
// Stage 2: banded attention, online softmax. One wave per 16-query tile.
// grid = (Wc/16, Cc, B*H), block = 32
// ---------------------------------------------------------------------------
__global__ __launch_bounds__(32)
void attn_kernel(const _Float16* __restrict__ qws,
                 const _Float16* __restrict__ kws,
                 const _Float16* __restrict__ vtws,
                 float* __restrict__ out)
{
  const int lane = threadIdx.x & 31;
  const int lo = lane & 15, hi = lane >> 4;
  const int qt = blockIdx.x;
  const int c  = blockIdx.y;
  const int bh = blockIdx.z;
  const int b = bh / Hv, h = bh - b * Hv;

  __shared__ _Float16 pbuf[16 * 32];

  const int qi0 = qt << 4;
  const _Float16* __restrict__ qptr =
      qws + ((size_t)bh * Sv + (c * Wc + qi0 + lo)) * Dv + hi * 8;

  V16 qa0, qa1;                          // q A-frags for D chunks [0,32),[32,64)
  qa0.h[0] = *(const v8h*)(qptr);        // k = hi*8 + 0..7
  qa0.h[1] = *(const v8h*)(qptr + 16);   // k = 16 + hi*8 + 0..7
  qa1.h[0] = *(const v8h*)(qptr + 32);
  qa1.h[1] = *(const v8h*)(qptr + 48);

  v8f o0 = {}, o1 = {}, o2 = {}, o3 = {};
  float mrow[8], lrow[8];
#pragma unroll
  for (int r = 0; r < 8; ++r) { mrow[r] = -1e30f; lrow[r] = 0.0f; }

  const int kgbase = (c - 1) * Wc;

  for (int kk = 0; kk < 3 * Wc; kk += 32) {
    // ---- QK^T: two 16x16 score tiles
    v8f s0 = {}, s1 = {};
#pragma unroll
    for (int t = 0; t < 2; ++t) {
      const int keyg = kgbase + kk + t * 16 + lo;
      V16 b0, b1;
      if (keyg >= 0 && keyg < Sv) {
        const _Float16* kp = kws + ((size_t)bh * Sv + keyg) * Dv + hi * 8;
        b0.h[0] = *(const v8h*)(kp);       // d = hi*8 + 0..7, 16 + hi*8 + 0..7
        b0.h[1] = *(const v8h*)(kp + 16);
        b1.h[0] = *(const v8h*)(kp + 32);
        b1.h[1] = *(const v8h*)(kp + 48);
      } else {
        b0.v = (v16h)(_Float16)0.0f; b1.v = (v16h)(_Float16)0.0f;
      }
      v8f sc = {};
      sc = wmma_f16(qa0.v, b0.v, sc);
      sc = wmma_f16(qa1.v, b1.v, sc);
      if (t == 0) s0 = sc; else s1 = sc;
    }

    // ---- band mask + online softmax
    float p0[8], p1[8];
#pragma unroll
    for (int r = 0; r < 8; ++r) {
      const int qi = qi0 + r + hi * 8;
      const int ka0 = kk + lo, ka1 = kk + 16 + lo;
      const bool v0 = (ka0 >= qi) && (ka0 <= qi + 2 * Wc) &&
                      (kgbase + ka0 >= 0) && (kgbase + ka0 < Sv);
      const bool v1 = (ka1 >= qi) && (ka1 <= qi + 2 * Wc) &&
                      (kgbase + ka1 >= 0) && (kgbase + ka1 < Sv);
      float x0 = v0 ? s0[r] : -1e30f;
      float x1 = v1 ? s1[r] : -1e30f;
      float mx = fmaxf(x0, x1);
#pragma unroll
      for (int off = 1; off < 16; off <<= 1)
        mx = fmaxf(mx, __shfl_xor(mx, off, 32));
      const float mn = fmaxf(mrow[r], mx);
      const float corr = __expf(mrow[r] - mn);
      p0[r] = v0 ? __expf(x0 - mn) : 0.0f;
      p1[r] = v1 ? __expf(x1 - mn) : 0.0f;
      float rs = p0[r] + p1[r];
#pragma unroll
      for (int off = 1; off < 16; off <<= 1)
        rs += __shfl_xor(rs, off, 32);
      lrow[r] = lrow[r] * corr + rs;
      mrow[r] = mn;
      o0[r] *= corr; o1[r] *= corr; o2[r] *= corr; o3[r] *= corr;
    }

    // ---- P tile: C-layout -> LDS -> A-layout fragment
#pragma unroll
    for (int r = 0; r < 8; ++r) {
      pbuf[(r + hi * 8) * 32 + lo]      = (_Float16)p0[r];
      pbuf[(r + hi * 8) * 32 + 16 + lo] = (_Float16)p1[r];
    }
    __syncthreads();
    V16 pa;
    {
      const _Float16* pp = &pbuf[lo * 32 + hi * 16];
      pa.h[0] = *(const v8h*)(pp);
      pa.h[1] = *(const v8h*)(pp + 8);
    }
    __syncthreads();

    // ---- PV: 4 output column tiles (D=64)
#pragma unroll
    for (int t = 0; t < 4; ++t) {
      const _Float16* vp = vtws + ((size_t)bh * Dv + t * 16 + lo) * Sv;
      const int kb = kgbase + kk + hi * 16;
      V16 vb;
      if (kb >= 0 && kb + 16 <= Sv) {
        vb.h[0] = *(const v8h*)(vp + kb);
        vb.h[1] = *(const v8h*)(vp + kb + 8);
      } else {
#pragma unroll
        for (int e = 0; e < 16; ++e) {
          const int kg = kb + e;
          vb.v[e] = (kg >= 0 && kg < Sv) ? vp[kg] : (_Float16)0.0f;
        }
      }
      v8f acc = (t == 0) ? o0 : (t == 1) ? o1 : (t == 2) ? o2 : o3;
      acc = wmma_f16(pa.v, vb.v, acc);
      if (t == 0) o0 = acc; else if (t == 1) o1 = acc;
      else if (t == 2) o2 = acc; else o3 = acc;
    }
  }

  // ---- epilogue
#pragma unroll
  for (int r = 0; r < 8; ++r) {
    const float inv = 1.0f / lrow[r];
    const int s = c * Wc + qi0 + r + hi * 8;
    float* op = out + ((size_t)b * Sv + s) * Ev + h * Dv;
    op[0 * 16 + lo] = o0[r] * inv;
    op[1 * 16 + lo] = o1[r] * inv;
    op[2 * 16 + lo] = o2[r] * inv;
    op[3 * 16 + lo] = o3[r] * inv;
  }
}

// ---------------------------------------------------------------------------
extern "C" void kernel_launch(void* const* d_in, const int* in_sizes, int n_in,
                              void* d_out, int out_size, void* d_ws, size_t ws_size,
                              hipStream_t stream) {
  (void)in_sizes; (void)n_in; (void)out_size; (void)ws_size;
  const float* X  = (const float*)d_in[0];
  const float* Wq = (const float*)d_in[1];
  const float* bq = (const float*)d_in[2];
  const float* Wk = (const float*)d_in[3];
  const float* bk = (const float*)d_in[4];
  const float* Wv = (const float*)d_in[5];
  const float* bv = (const float*)d_in[6];

  const size_t nX = (size_t)Bv * Sv * Ev;          // 6,291,456
  const size_t nW = (size_t)3 * Ev * Ev;           // 1,769,472
  const size_t nE = (size_t)Bv * Hv * Sv * Dv;     // 6,291,456

  _Float16* Xh   = (_Float16*)d_ws;
  _Float16* WhT  = Xh + nX;
  _Float16* qws  = WhT + nW;
  _Float16* kws  = qws + nE;
  _Float16* vtws = kws + nE;                       // total ~54 MB

  cvt_x_kernel<<<dim3((unsigned)((nX + 255) / 256)), 256, 0, stream>>>(X, Xh, (int)nX);
  cvt_w_kernel<<<dim3((unsigned)((nW + 255) / 256)), 256, 0, stream>>>(Wq, Wk, Wv, WhT);

  dim3 gA((Bv * Sv) / 64, Ev / 64, 3);
  qkv_gemm_kernel<<<gA, 128, 0, stream>>>(Xh, WhT, bq, bk, bv, qws, kws, vtws);

  dim3 gB(Wc / 16, Cc, Bv * Hv);
  attn_kernel<<<gB, 32, 0, stream>>>(qws, kws, vtws, (float*)d_out);
}
